// TransformerModel_88072599372192
// MI455X (gfx1250) — compile-verified, matrix operand order checked by reading, not run on hardware
//
#include <hip/hip_runtime.h>

// ---------------------------------------------------------------------------
// Transformer (enc-dec) forward for gfx1250 using v_wmma_f32_16x16x32_f16.
// All GEMMs (QKV/attn/out/FFN/logits) go through one WMMA tile kernel with
// double-buffered LDS staging (f32 global -> f16 LDS) and vectorized b128
// loads. All M,N are multiples of 64 and K multiples of 32 in this model, so
// staging needs no bounds checks.
// ---------------------------------------------------------------------------

typedef __attribute__((ext_vector_type(16))) _Float16 v16h;
typedef __attribute__((ext_vector_type(4)))  _Float16 v4h;
typedef __attribute__((ext_vector_type(8)))  float    v8f;

constexpr int NLAYER = 6;
constexpr int DMODEL = 512;
constexpr int NHEAD  = 8;
constexpr int DHEAD  = 64;   // DK == DV
constexpr int DFFN   = 2048;
constexpr int NVOCAB = 32000;
constexpr int NBATCH = 4;
constexpr int LSEQ   = 512;  // LS == LT

// ---------------------------------------------------------------------------
// WMMA fragment load from an LDS tile stored row-major [row][32] f16.
// 16-bit A-matrix 16x32 layout (ISA 7.12.2): lane&15 = M row,
// element e -> K = 16*(e>>3) + 8*(lane>>4) + (e&7).
// B tiles are staged transposed ([n][k]) so the same loader works for B.
// Consecutive even/odd e are K-adjacent -> compiler fuses into ds_load_b128.
// ---------------------------------------------------------------------------
__device__ __forceinline__ v16h load_frag_lds(const _Float16* tile, int lane, int rowBase)
{
    const _Float16* p = tile + (rowBase + (lane & 15)) * 32;
    const int kb = (lane >> 4) << 3;   // 0 or 8
    v16h f;
#pragma unroll
    for (int e = 0; e < 16; ++e) {
        const int k = ((e & 8) << 1) + kb + (e & 7);
        f[e] = p[k];
    }
    return f;
}

// ---------------------------------------------------------------------------
// Generic batched GEMM:  C = alpha * (A @ B) + bias [, ReLU]
//   A: [M,K] row-major, row stride lda, batch offset bb*sAb + hh*sAh
//   B: !TRANSB -> [K,N] (ldb = row stride); TRANSB -> [N,K] (ldb = row stride)
//   C: [M,N], ldc
// batch index z = blockIdx.z, bb = z / batchH, hh = z % batchH.
// Block = 128 threads (4 waves), block tile 64x64, wave tile 32x32, K step 32.
// Requires: M%64==0 is NOT required (epilogue guards), but staging assumes the
// grid exactly tiles M,N and K%32==0 -- true for every GEMM in this model.
// ---------------------------------------------------------------------------
template <bool TRANSB, bool RELU>
__global__ __launch_bounds__(128) void gemm_kernel(
    const float* __restrict__ A, const float* __restrict__ Bm,
    const float* __restrict__ bias, float* __restrict__ C,
    int M, int N, int K, int lda, int ldb, int ldc, int batchH,
    long long sAb, long long sAh, long long sBb, long long sBh,
    long long sCb, long long sCh, float alpha)
{
    __shared__ _Float16 As[2][64 * 32];
    __shared__ _Float16 Bs[2][64 * 32];   // stored as [n][k]

    const int z  = blockIdx.z;
    const int bb = z / batchH;
    const int hh = z % batchH;
    const float* __restrict__ Ab = A  + bb * sAb + hh * sAh;
    const float* __restrict__ Bb = Bm + bb * sBb + hh * sBh;
    float*       __restrict__ Cb = C  + bb * sCb + hh * sCh;

    const int row0 = blockIdx.y * 64;
    const int col0 = blockIdx.x * 64;
    const int tid  = threadIdx.x;
    const int lane = tid & 31;
    const int wave = tid >> 5;
    const int wm   = (wave >> 1) * 32;   // wave row offset in block tile
    const int wn   = (wave & 1) * 32;    // wave col offset in block tile

    v8f acc[2][2] = {};
    float4 ar[4], br[4];

    // ---- vectorized tile load: 4 x float4 per operand per thread ---------
    auto load_tile = [&](int k0) {
#pragma unroll
        for (int i = 0; i < 4; ++i) {
            const int j  = i * 128 + tid;      // float4 index in 64x32 tile
            const int ra = j >> 3;             // row (A / B-trans): 8 float4 per row
            const int ca = (j & 7) << 2;       // k offset
            ar[i] = *(const float4*)(Ab + (long long)(row0 + ra) * lda + (k0 + ca));
            if (TRANSB) {
                br[i] = *(const float4*)(Bb + (long long)(col0 + ra) * ldb + (k0 + ca));
            } else {
                const int kr = j >> 4;         // k row: 16 float4 per 64-wide row
                const int cn = (j & 15) << 2;  // n offset
                br[i] = *(const float4*)(Bb + (long long)(k0 + kr) * ldb + (col0 + cn));
            }
        }
    };

    // ---- convert + store staged registers into LDS buffer ----------------
    auto store_tile = [&](int buf) {
#pragma unroll
        for (int i = 0; i < 4; ++i) {
            const int j  = i * 128 + tid;
            const int ra = j >> 3;
            const int ca = (j & 7) << 2;
            v4h av = { (_Float16)ar[i].x, (_Float16)ar[i].y,
                       (_Float16)ar[i].z, (_Float16)ar[i].w };
            *(v4h*)(&As[buf][ra * 32 + ca]) = av;
            if (TRANSB) {
                v4h bv = { (_Float16)br[i].x, (_Float16)br[i].y,
                           (_Float16)br[i].z, (_Float16)br[i].w };
                *(v4h*)(&Bs[buf][ra * 32 + ca]) = bv;
            } else {
                const int kr = j >> 4;
                const int cn = (j & 15) << 2;
                Bs[buf][(cn + 0) * 32 + kr] = (_Float16)br[i].x;
                Bs[buf][(cn + 1) * 32 + kr] = (_Float16)br[i].y;
                Bs[buf][(cn + 2) * 32 + kr] = (_Float16)br[i].z;
                Bs[buf][(cn + 3) * 32 + kr] = (_Float16)br[i].w;
            }
        }
    };

    const int nk = K >> 5;  // K / 32, exact
    load_tile(0);
    store_tile(0);
    int cur = 0;

    for (int s = 0; s < nk; ++s) {
        __syncthreads();
        const bool more = (s + 1) < nk;
        if (more) load_tile((s + 1) << 5);   // overlap with WMMA below
        if (s + 2 < nk) {                    // gfx1250 global_prefetch path
            const int kp = (s + 2) << 5;
            __builtin_prefetch(Ab + (long long)(row0 + (tid >> 3)) * lda + kp, 0, 0);
            if (TRANSB)
                __builtin_prefetch(Bb + (long long)(col0 + (tid >> 3)) * ldb + kp, 0, 0);
            else
                __builtin_prefetch(Bb + (long long)(kp + (tid >> 4)) * ldb + col0, 0, 0);
        }

        const _Float16* Al = As[cur];
        const _Float16* Bl = Bs[cur];
        const v16h a0 = load_frag_lds(Al, lane, wm);
        const v16h a1 = load_frag_lds(Al, lane, wm + 16);
        const v16h b0 = load_frag_lds(Bl, lane, wn);
        const v16h b1 = load_frag_lds(Bl, lane, wn + 16);

        acc[0][0] = __builtin_amdgcn_wmma_f32_16x16x32_f16(false, a0, false, b0, (short)0, acc[0][0], false, false);
        acc[0][1] = __builtin_amdgcn_wmma_f32_16x16x32_f16(false, a0, false, b1, (short)0, acc[0][1], false, false);
        acc[1][0] = __builtin_amdgcn_wmma_f32_16x16x32_f16(false, a1, false, b0, (short)0, acc[1][0], false, false);
        acc[1][1] = __builtin_amdgcn_wmma_f32_16x16x32_f16(false, a1, false, b1, (short)0, acc[1][1], false, false);

        if (more) store_tile(cur ^ 1);
        cur ^= 1;
    }

    // C/D 16x16 f32 layout: VGPR r -> M = r + 8*(lane>>4), N = lane&15
    const int rsub = (lane >> 4) << 3;
    const int csub = lane & 15;
#pragma unroll
    for (int i = 0; i < 2; ++i)
#pragma unroll
        for (int j = 0; j < 2; ++j)
#pragma unroll
            for (int r = 0; r < 8; ++r) {
                const int m = row0 + wm + i * 16 + rsub + r;
                const int n = col0 + wn + j * 16 + csub;
                if (m < M && n < N) {
                    float v = alpha * acc[i][j][r];
                    if (bias) v += bias[n];
                    if (RELU) v = fmaxf(v, 0.0f);
                    Cb[(long long)m * ldc + n] = v;
                }
            }
}

// ---------------------------------------------------------------------------
// Row softmax over scores [BH, Lq, Lk] with optional causal mask (col>q -> -1e9)
// ---------------------------------------------------------------------------
__global__ __launch_bounds__(256) void softmax_kernel(float* __restrict__ s,
                                                      int Lq, int Lk, int causal)
{
    const int q = blockIdx.x;
    float* row = s + ((long long)blockIdx.y * Lq + q) * Lk;
    __shared__ float red[256];
    const int tid = threadIdx.x;

    float m = -3.0e38f;
    for (int c = tid; c < Lk; c += 256) {
        float v = row[c];
        if (causal && c > q) v = -1.0e9f;
        m = fmaxf(m, v);
    }
    red[tid] = m; __syncthreads();
    for (int o = 128; o > 0; o >>= 1) { if (tid < o) red[tid] = fmaxf(red[tid], red[tid + o]); __syncthreads(); }
    m = red[0]; __syncthreads();

    float sum = 0.0f;
    for (int c = tid; c < Lk; c += 256) {
        float v = row[c];
        if (causal && c > q) v = -1.0e9f;
        const float e = expf(v - m);
        row[c] = e;
        sum += e;
    }
    red[tid] = sum; __syncthreads();
    for (int o = 128; o > 0; o >>= 1) { if (tid < o) red[tid] += red[tid + o]; __syncthreads(); }
    const float inv = 1.0f / red[0];
    for (int c = tid; c < Lk; c += 256) row[c] *= inv;
}

// ---------------------------------------------------------------------------
// out = LayerNorm(x [+ res]) * g + b   (row length DMODEL = 512, block 128)
// ---------------------------------------------------------------------------
__global__ __launch_bounds__(128) void ln_kernel(const float* __restrict__ x,
                                                 const float* __restrict__ res,
                                                 const float* __restrict__ g,
                                                 const float* __restrict__ b,
                                                 float* __restrict__ out)
{
    const long long row = blockIdx.x;
    const float* xr = x + row * DMODEL;
    const float* rr = res ? res + row * DMODEL : nullptr;
    const int tid = threadIdx.x;
    __shared__ float red[128];

    float v[4];
    float s = 0.0f;
#pragma unroll
    for (int i = 0; i < 4; ++i) {
        const int c = tid + i * 128;
        float t = xr[c];
        if (rr) t += rr[c];
        v[i] = t;
        s += t;
    }
    red[tid] = s; __syncthreads();
    for (int o = 64; o > 0; o >>= 1) { if (tid < o) red[tid] += red[tid + o]; __syncthreads(); }
    const float mu = red[0] / (float)DMODEL;
    __syncthreads();

    float s2 = 0.0f;
#pragma unroll
    for (int i = 0; i < 4; ++i) { const float d = v[i] - mu; s2 += d * d; }
    red[tid] = s2; __syncthreads();
    for (int o = 64; o > 0; o >>= 1) { if (tid < o) red[tid] += red[tid + o]; __syncthreads(); }
    const float inv = rsqrtf(red[0] / (float)DMODEL + 1e-5f);

#pragma unroll
    for (int i = 0; i < 4; ++i) {
        const int c = tid + i * 128;
        out[row * DMODEL + c] = (v[i] - mu) * inv * g[c] + b[c];
    }
}

// ---------------------------------------------------------------------------
// x[b,l,:] = embed[ids[b,l],:] * sqrt(D) + PE[l,:]   (PE computed on the fly)
// ---------------------------------------------------------------------------
__global__ __launch_bounds__(256) void embed_kernel(const int* __restrict__ ids,
                                                    const float* __restrict__ emb,
                                                    float* __restrict__ out, int L)
{
    const int l = blockIdx.x;
    const int bb = blockIdx.y;
    const int token = ids[bb * L + l];
    const float scale = sqrtf((float)DMODEL);
    for (int c = threadIdx.x; c < DMODEL; c += 256) {
        const int i2 = (c >> 1) << 1;  // 2*floor(c/2)
        const float div = expf(-logf(10000.0f) * (float)i2 / (float)DMODEL);
        const float arg = (float)l * div;
        const float pe = (c & 1) ? cosf(arg) : sinf(arg);
        out[((long long)bb * L + l) * DMODEL + c] = emb[(long long)token * DMODEL + c] * scale + pe;
    }
}

// ---------------------------------------------------------------------------
// Host-side orchestration
// ---------------------------------------------------------------------------
struct AttnW { const float *wq, *bq, *wk, *bk, *wv, *bv, *wo, *bo; };
struct FfnW  { const float *w1, *b1, *w2, *b2; };

// JAX tree-flatten order within an attn dict: bk bo bq bv wk wo wq wv
static AttnW attn_from(void* const* d_in, int base) {
    AttnW a;
    a.bk = (const float*)d_in[base + 0]; a.bo = (const float*)d_in[base + 1];
    a.bq = (const float*)d_in[base + 2]; a.bv = (const float*)d_in[base + 3];
    a.wk = (const float*)d_in[base + 4]; a.wo = (const float*)d_in[base + 5];
    a.wq = (const float*)d_in[base + 6]; a.wv = (const float*)d_in[base + 7];
    return a;
}
// ffn dict: b1 b2 w1 w2
static FfnW ffn_from(void* const* d_in, int base) {
    FfnW f;
    f.b1 = (const float*)d_in[base + 0]; f.b2 = (const float*)d_in[base + 1];
    f.w1 = (const float*)d_in[base + 2]; f.w2 = (const float*)d_in[base + 3];
    return f;
}

static void run_gemm(hipStream_t st, const float* A, const float* Bm, const float* bias,
                     float* C, int M, int N, int K, int lda, int ldb, int ldc,
                     int batch, int batchH,
                     long long sAb, long long sAh, long long sBb, long long sBh,
                     long long sCb, long long sCh,
                     float alpha, bool transB, bool relu)
{
    dim3 grid((N + 63) / 64, (M + 63) / 64, batch);
    dim3 block(128);
    if (transB) {
        if (relu) gemm_kernel<true, true ><<<grid, block, 0, st>>>(A, Bm, bias, C, M, N, K, lda, ldb, ldc, batchH, sAb, sAh, sBb, sBh, sCb, sCh, alpha);
        else      gemm_kernel<true, false><<<grid, block, 0, st>>>(A, Bm, bias, C, M, N, K, lda, ldb, ldc, batchH, sAb, sAh, sBb, sBh, sCb, sCh, alpha);
    } else {
        if (relu) gemm_kernel<false, true ><<<grid, block, 0, st>>>(A, Bm, bias, C, M, N, K, lda, ldb, ldc, batchH, sAb, sAh, sBb, sBh, sCb, sCh, alpha);
        else      gemm_kernel<false, false><<<grid, block, 0, st>>>(A, Bm, bias, C, M, N, K, lda, ldb, ldc, batchH, sAb, sAh, sBb, sBh, sCb, sCh, alpha);
    }
}

static void run_ln(hipStream_t st, const float* x, const float* res,
                   const float* g, const float* b, float* out, int rows)
{
    ln_kernel<<<rows, 128, 0, st>>>(x, res, g, b, out);
}

static void run_mha(hipStream_t st, const float* xq, const float* xkv, int Lq, int Lk,
                    const AttnW& p, int layer, bool causal,
                    float* Qb, float* Kb, float* Vb, float* Sb, float* Cb, float* Ob)
{
    const long long wS = (long long)DMODEL * NHEAD * DHEAD;
    const long long bS = NHEAD * DHEAD;
    const float* wq = p.wq + layer * wS;  const float* bq = p.bq + layer * bS;
    const float* wk = p.wk + layer * wS;  const float* bk = p.bk + layer * bS;
    const float* wv = p.wv + layer * wS;  const float* bv = p.bv + layer * bS;
    const float* wo = p.wo + (long long)layer * NHEAD * DHEAD * DMODEL;
    const float* bo = p.bo + (long long)layer * DMODEL;
    const int HD = NHEAD * DHEAD;  // 512
    const int Mq = NBATCH * Lq, Mk = NBATCH * Lk;

    // Q/K/V projections
    run_gemm(st, xq,  wq, bq, Qb, Mq, HD, DMODEL, DMODEL, HD, HD, 1, 1, 0,0,0,0,0,0, 1.0f, false, false);
    run_gemm(st, xkv, wk, bk, Kb, Mk, HD, DMODEL, DMODEL, HD, HD, 1, 1, 0,0,0,0,0,0, 1.0f, false, false);
    run_gemm(st, xkv, wv, bv, Vb, Mk, HD, DMODEL, DMODEL, HD, HD, 1, 1, 0,0,0,0,0,0, 1.0f, false, false);

    // scores[b,h,q,k] = (Q_bh @ K_bh^T) / sqrt(DHEAD)
    run_gemm(st, Qb, Kb, nullptr, Sb, Lq, Lk, DHEAD, HD, HD, Lk,
             NBATCH * NHEAD, NHEAD,
             (long long)Lq * HD, DHEAD,
             (long long)Lk * HD, DHEAD,
             (long long)NHEAD * Lq * Lk, (long long)Lq * Lk,
             0.125f, /*transB=*/true, false);

    softmax_kernel<<<dim3(Lq, NBATCH * NHEAD), 256, 0, st>>>(Sb, Lq, Lk, causal ? 1 : 0);

    // ctx[b,q,h,:] = attn_bh @ V_bh   (written directly into [B,Lq,H*DV] layout)
    run_gemm(st, Sb, Vb, nullptr, Cb, Lq, DHEAD, Lk, Lk, HD, HD,
             NBATCH * NHEAD, NHEAD,
             (long long)NHEAD * Lq * Lk, (long long)Lq * Lk,
             (long long)Lk * HD, DHEAD,
             (long long)Lq * HD, DHEAD,
             1.0f, false, false);

    // output projection
    run_gemm(st, Cb, wo, bo, Ob, Mq, DMODEL, HD, HD, DMODEL, DMODEL, 1, 1, 0,0,0,0,0,0, 1.0f, false, false);
}

static void run_ffn(hipStream_t st, const float* x, const FfnW& f, int layer,
                    float* hid, float* out)
{
    const float* w1 = f.w1 + (long long)layer * DMODEL * DFFN;
    const float* b1 = f.b1 + (long long)layer * DFFN;
    const float* w2 = f.w2 + (long long)layer * DFFN * DMODEL;
    const float* b2 = f.b2 + (long long)layer * DMODEL;
    const int TOK = NBATCH * LSEQ;
    run_gemm(st, x,   w1, b1, hid, TOK, DFFN,   DMODEL, DMODEL, DFFN,   DFFN,   1, 1, 0,0,0,0,0,0, 1.0f, false, /*relu=*/true);
    run_gemm(st, hid, w2, b2, out, TOK, DMODEL, DFFN,   DFFN,   DMODEL, DMODEL, 1, 1, 0,0,0,0,0,0, 1.0f, false, false);
}

extern "C" void kernel_launch(void* const* d_in, const int* in_sizes, int n_in,
                              void* d_out, int out_size, void* d_ws, size_t ws_size,
                              hipStream_t stream)
{
    (void)in_sizes; (void)n_in; (void)out_size; (void)ws_size;

    // --- input decoding (JAX tree-flatten, dict keys sorted) --------------
    // 0: src_ids  1: tgt_ids  2..: params flattened:
    //   dec: cross(8) ffn(4) ln1_b ln1_g ln2_b ln2_g ln3_b ln3_g lnf_b lnf_g self(8)
    //   embed
    //   enc: attn(8) ffn(4) ln1_b ln1_g ln2_b ln2_g lnf_b lnf_g
    //   out_b
    const int* src_ids = (const int*)d_in[0];
    const int* tgt_ids = (const int*)d_in[1];

    AttnW decCross = attn_from(d_in, 2);
    FfnW  decFfn   = ffn_from(d_in, 10);
    const float* dln1b = (const float*)d_in[14]; const float* dln1g = (const float*)d_in[15];
    const float* dln2b = (const float*)d_in[16]; const float* dln2g = (const float*)d_in[17];
    const float* dln3b = (const float*)d_in[18]; const float* dln3g = (const float*)d_in[19];
    const float* dlnfb = (const float*)d_in[20]; const float* dlnfg = (const float*)d_in[21];
    AttnW decSelf  = attn_from(d_in, 22);
    const float* embed = (const float*)d_in[30];
    AttnW encAttn  = attn_from(d_in, 31);
    FfnW  encFfn   = ffn_from(d_in, 39);
    const float* eln1b = (const float*)d_in[43]; const float* eln1g = (const float*)d_in[44];
    const float* eln2b = (const float*)d_in[45]; const float* eln2g = (const float*)d_in[46];
    const float* elnfb = (const float*)d_in[47]; const float* elnfg = (const float*)d_in[48];
    const float* out_b = (const float*)d_in[49];

    // --- workspace carve (floats) ----------------------------------------
    const int TOK = NBATCH * LSEQ;  // 2048
    float* w = (float*)d_ws;
    size_t off = 0;
    auto alloc = [&](size_t n) { float* p = w + off; off += n; return p; };
    float* X   = alloc((size_t)TOK * DMODEL);                      // activations
    float* T1  = alloc((size_t)TOK * DMODEL);                      // sublayer output
    float* Qb  = alloc((size_t)TOK * NHEAD * DHEAD);
    float* Kb  = alloc((size_t)TOK * NHEAD * DHEAD);
    float* Vb  = alloc((size_t)TOK * NHEAD * DHEAD);
    float* Sb  = alloc((size_t)NBATCH * NHEAD * LSEQ * LSEQ);      // scores
    float* Cb  = alloc((size_t)TOK * NHEAD * DHEAD);               // attn context
    float* HID = alloc((size_t)TOK * DFFN);                        // ffn hidden
    float* ENC = alloc((size_t)TOK * DMODEL);                      // encoder output
    float* Y   = alloc((size_t)TOK * DMODEL);                      // decoder activations

    // --- encoder ----------------------------------------------------------
    embed_kernel<<<dim3(LSEQ, NBATCH), 256, 0, stream>>>(src_ids, embed, X, LSEQ);
    for (int i = 0; i < NLAYER; ++i) {
        run_mha(stream, X, X, LSEQ, LSEQ, encAttn, i, false, Qb, Kb, Vb, Sb, Cb, T1);
        run_ln(stream, X, T1, eln1g + i * DMODEL, eln1b + i * DMODEL, X, TOK);
        run_ffn(stream, X, encFfn, i, HID, T1);
        run_ln(stream, X, T1, eln2g + i * DMODEL, eln2b + i * DMODEL, X, TOK);
    }
    run_ln(stream, X, nullptr, elnfg, elnfb, ENC, TOK);

    // --- decoder ----------------------------------------------------------
    embed_kernel<<<dim3(LSEQ, NBATCH), 256, 0, stream>>>(tgt_ids, embed, Y, LSEQ);
    for (int i = 0; i < NLAYER; ++i) {
        run_mha(stream, Y, Y, LSEQ, LSEQ, decSelf, i, /*causal=*/true, Qb, Kb, Vb, Sb, Cb, T1);
        run_ln(stream, Y, T1, dln1g + i * DMODEL, dln1b + i * DMODEL, Y, TOK);
        run_mha(stream, Y, ENC, LSEQ, LSEQ, decCross, i, false, Qb, Kb, Vb, Sb, Cb, T1);
        run_ln(stream, Y, T1, dln2g + i * DMODEL, dln2b + i * DMODEL, Y, TOK);
        run_ffn(stream, Y, decFfn, i, HID, T1);
        run_ln(stream, Y, T1, dln3g + i * DMODEL, dln3b + i * DMODEL, Y, TOK);
    }
    run_ln(stream, Y, nullptr, dlnfg, dlnfb, Y, TOK);

    // --- tied output projection: logits = Y @ embed^T + out_b ------------
    run_gemm(stream, Y, embed, out_b, (float*)d_out,
             TOK, NVOCAB, DMODEL, DMODEL, DMODEL, NVOCAB,
             1, 1, 0, 0, 0, 0, 0, 0, 1.0f, /*transB=*/true, false);
}
